// VisionAttention_20976620273967
// MI455X (gfx1250) — compile-verified, matrix operand order checked by reading, not run on hardware
//
#include <hip/hip_runtime.h>
#include <hip/hip_bf16.h>

typedef __attribute__((ext_vector_type(16))) __bf16 v16bf;
typedef __attribute__((ext_vector_type(8)))  __bf16 v8bf;
typedef __attribute__((ext_vector_type(4)))  __bf16 v4bf;
typedef __attribute__((ext_vector_type(8)))  float  v8f;

#define Bsz 8
#define Nsq 1024
#define Dm  768
#define Hh  12
#define DH  64
#define WSZ (Dm * Dm)   // one weight matrix, elems

// 16-bit fragment: lane ln = row(M)/col(N); per-lane elements are two
// contiguous K-runs of 8: [8*hi, 8*hi+8) and [16+8*hi, 16+8*hi+8).
// p must point at (base K-chunk + 8*hi), 16-byte aligned.
__device__ __forceinline__ v16bf load_frag(const __bf16* p) {
    v8bf lo = *(const v8bf*)p;
    v8bf hi = *(const v8bf*)(p + 16);
    return __builtin_shufflevector(lo, hi, 0, 1, 2, 3, 4, 5, 6, 7,
                                   8, 9, 10, 11, 12, 13, 14, 15);
}

// ---------------------------------------------------------------------------
// Stage 0: f32 -> bf16 pack (4 elems/thread, float4 in, 8B out)
// ---------------------------------------------------------------------------
__global__ void pack_bf16_kernel(const float* __restrict__ src,
                                 __bf16* __restrict__ dst, int n4)
{
    int i = blockIdx.x * blockDim.x + threadIdx.x;
    if (i < n4) {
        float4 v = ((const float4*)src)[i];
        v4bf o = { (__bf16)v.x, (__bf16)v.y, (__bf16)v.z, (__bf16)v.w };
        ((v4bf*)dst)[i] = o;
    }
}

// ---------------------------------------------------------------------------
// Stage 1: QKV projection (bf16 WMMA, f32 acc) + RMSNorm + 2D RoPE.
// grid = (BN/16, H, 3 {q,k,v}), block = 128 (4 waves).
// q,k -> [B,H,N,DH] bf16 ; v -> transposed [B,H,DH,N] bf16.
// ---------------------------------------------------------------------------
__global__ void qkv_norm_rope_kernel(const __bf16* __restrict__ hsb,
                                     const __bf16* __restrict__ Wb,  // q,k,v packed
                                     const int*    __restrict__ pos_ids,
                                     const float*  __restrict__ qscale,
                                     const float*  __restrict__ kscale,
                                     __bf16* __restrict__ qb,
                                     __bf16* __restrict__ kb,
                                     __bf16* __restrict__ vt)
{
    const int m    = blockIdx.z;           // 0=q 1=k 2=v
    const int h    = blockIdx.y;
    const int row0 = blockIdx.x * 16;      // token tile over B*N
    const int t    = threadIdx.x;
    const int w    = t >> 5;
    const int l    = t & 31;
    const int ln   = l & 15;
    const int hi   = l >> 4;
    const int col0 = h * DH + w * 16;

    __shared__ float tile[16][DH];
    __shared__ float rinv[16];

    const __bf16* arow = hsb + (size_t)(row0 + ln) * Dm + 8 * hi;
    const __bf16* brow = Wb + (size_t)m * WSZ + (size_t)(col0 + ln) * Dm + 8 * hi;

    v8f c = {};
    for (int kk = 0; kk < Dm; kk += 32) {
        __builtin_prefetch(arow + kk + 128, 0, 3);
        v16bf a = load_frag(arow + kk);
        v16bf b = load_frag(brow + kk);
        c = __builtin_amdgcn_wmma_f32_16x16x32_bf16(false, a, false, b,
                                                    (short)0, c, false, false);
    }
#pragma unroll
    for (int r = 0; r < 8; ++r)
        tile[r + 8 * hi][w * 16 + ln] = c[r];
    __syncthreads();

    // RMS norm (f32, eps=1e-6) per 64-wide row
    if (t < 16) {
        float s = 0.f;
        for (int d = 0; d < DH; ++d) { float x = tile[t][d]; s += x * x; }
        rinv[t] = rsqrtf(s * (1.0f / DH) + 1e-6f);
    }
    __syncthreads();
    for (int idx = t; idx < 16 * DH; idx += 128) {
        int r = idx >> 6, d = idx & 63;
        float sc = (m == 0) ? qscale[d] : (m == 1) ? kscale[d] : 1.0f;
        tile[r][d] = tile[r][d] * rinv[r] * sc;
    }
    __syncthreads();

    // 2D RoPE (q,k only)
    for (int idx = t; idx < 16 * DH; idx += 128) {
        int r = idx >> 6, d = idx & 63;
        int tok = row0 + r;
        int bidx = tok >> 10, n = tok & (Nsq - 1);
        float val = tile[r][d];
        if (m < 2) {
            int kc = d >> 5, j = d & 31, f = j & 15;
            float p   = (float)pos_ids[tok * 2 + kc];
            float ang = p * __expf(-(float)f * 0.28782313662425572f); // ln(100)/16
            float cs = __cosf(ang), sn = __sinf(ang);
            int dd = (j < 16) ? (d + 16) : (d - 16);
            float rot = (j < 16) ? -tile[r][dd] : tile[r][dd];
            val = val * cs + rot * sn;
            size_t o = (((size_t)bidx * Hh + h) * Nsq + n) * DH + d;
            ((m == 0) ? qb : kb)[o] = (__bf16)val;
        } else {
            // transposed: [B,H,DH,N]
            size_t o = (((size_t)bidx * Hh + h) * DH + d) * Nsq + n;
            vt[o] = (__bf16)val;
        }
    }
}

// ---------------------------------------------------------------------------
// Stage 2: single-pass flash attention. grid = (N/16, H, B), block = 128.
// Wave w computes S for kv tile kv0+16w and owns output cols [16w,16w+16).
// Block-wide online softmax merged through LDS each 64-kv step.
// ---------------------------------------------------------------------------
__global__ void attn_kernel(const __bf16* __restrict__ qb,
                            const __bf16* __restrict__ kb,
                            const __bf16* __restrict__ vt,
                            __bf16* __restrict__ attnb)
{
    const int q0 = blockIdx.x * 16;
    const int h  = blockIdx.y;
    const int b  = blockIdx.z;
    const int t  = threadIdx.x;
    const int w  = t >> 5;
    const int l  = t & 31;
    const int ln = l & 15;
    const int hi = l >> 4;
    const size_t base  = ((size_t)b * Hh + h) * Nsq * DH;   // q/k rows
    const size_t baseT = ((size_t)b * Hh + h) * DH * Nsq;   // v transposed

    __shared__ __align__(16) __bf16 ldsP[16][64];
    __shared__ float lds_m[4][16], lds_s[4][16];

    const __bf16* qrow = qb + base + (size_t)(q0 + ln) * DH + 8 * hi;
    v16bf qa[2];
#pragma unroll
    for (int cch = 0; cch < 2; ++cch)
        qa[cch] = load_frag(qrow + 32 * cch);

    float m_run[8], s_run[8];
#pragma unroll
    for (int r = 0; r < 8; ++r) { m_run[r] = -1e30f; s_run[r] = 0.f; }

    v8f acc = {};
    for (int kv0 = 0; kv0 < Nsq; kv0 += 64) {
        // --- S tile for this wave's 16 kv rows ---
        const __bf16* krow = kb + base + (size_t)(kv0 + w * 16 + ln) * DH + 8 * hi;
        __builtin_prefetch(krow + 64 * DH, 0, 3);
        v8f c = {};
#pragma unroll
        for (int cch = 0; cch < 2; ++cch) {
            v16bf bf = load_frag(krow + 32 * cch);
            c = __builtin_amdgcn_wmma_f32_16x16x32_bf16(false, qa[cch], false, bf,
                                                        (short)0, c, false, false);
        }
        // per-wave tile row max (reduce across 16 lanes of each half)
        float tmax[8];
#pragma unroll
        for (int r = 0; r < 8; ++r) {
            float v = c[r];
            for (int mk = 1; mk < 16; mk <<= 1) v = fmaxf(v, __shfl_xor(v, mk, 32));
            tmax[r] = v;
        }
        __syncthreads();                       // prior iter done reading LDS
        if (ln == 0) {
#pragma unroll
            for (int r = 0; r < 8; ++r) lds_m[w][hi * 8 + r] = tmax[r];
        }
        __syncthreads();
        // block-wide new running max
        float mnew[8];
#pragma unroll
        for (int r = 0; r < 8; ++r) {
            int row = r + 8 * hi;
            float tm = lds_m[0][row];
            tm = fmaxf(tm, lds_m[1][row]);
            tm = fmaxf(tm, lds_m[2][row]);
            tm = fmaxf(tm, lds_m[3][row]);
            mnew[r] = fmaxf(m_run[r], tm);
        }
        // P = exp(S - mnew) -> LDS; per-wave row sums -> LDS
#pragma unroll
        for (int r = 0; r < 8; ++r) {
            float e0 = __expf(c[r] - mnew[r]);
            ldsP[r + 8 * hi][w * 16 + ln] = (__bf16)e0;
            for (int mk = 1; mk < 16; mk <<= 1) e0 += __shfl_xor(e0, mk, 32);
            if (ln == 0) lds_s[w][hi * 8 + r] = e0;
        }
        __syncthreads();
        // online update: rescale acc & sum, then O += P @ V
#pragma unroll
        for (int r = 0; r < 8; ++r) {
            int row = r + 8 * hi;
            float sc = __expf(m_run[r] - mnew[r]);
            float ts = lds_s[0][row] + lds_s[1][row] + lds_s[2][row] + lds_s[3][row];
            s_run[r] = s_run[r] * sc + ts;
            acc[r]  *= sc;
            m_run[r] = mnew[r];
        }
        const __bf16* vrow = vt + baseT + (size_t)(w * 16 + ln) * Nsq + kv0 + 8 * hi;
        __builtin_prefetch(vrow + 64, 0, 3);
#pragma unroll
        for (int cch = 0; cch < 2; ++cch) {
            v16bf pa = load_frag(&ldsP[ln][8 * hi + 32 * cch]);
            v16bf vf = load_frag(vrow + 32 * cch);
            acc = __builtin_amdgcn_wmma_f32_16x16x32_bf16(false, pa, false, vf,
                                                          (short)0, acc, false, false);
        }
    }

#pragma unroll
    for (int r = 0; r < 8; ++r) {
        int row = r + 8 * hi;
        size_t tok = (size_t)b * Nsq + q0 + row;
        attnb[tok * Dm + h * DH + w * 16 + ln] = (__bf16)(acc[r] / s_run[r]);
    }
}

// ---------------------------------------------------------------------------
// Stage 3: output projection attnb(8192x768 bf16) @ Wo^T -> d_out (f32)
// ---------------------------------------------------------------------------
__global__ void out_proj_kernel(const __bf16* __restrict__ attnb,
                                const __bf16* __restrict__ Wob,
                                float* __restrict__ out)
{
    const int row0 = blockIdx.x * 16;
    const int col0 = blockIdx.y * 64 + (threadIdx.x >> 5) * 16;
    const int l  = threadIdx.x & 31;
    const int ln = l & 15;
    const int hi = l >> 4;

    const __bf16* arow = attnb + (size_t)(row0 + ln) * Dm + 8 * hi;
    const __bf16* brow = Wob   + (size_t)(col0 + ln) * Dm + 8 * hi;

    v8f c = {};
    for (int kk = 0; kk < Dm; kk += 32) {
        __builtin_prefetch(arow + kk + 128, 0, 3);
        v16bf a = load_frag(arow + kk);
        v16bf b = load_frag(brow + kk);
        c = __builtin_amdgcn_wmma_f32_16x16x32_bf16(false, a, false, b,
                                                    (short)0, c, false, false);
    }
#pragma unroll
    for (int r = 0; r < 8; ++r)
        out[(size_t)(row0 + r + 8 * hi) * Dm + col0 + ln] = c[r];
}

extern "C" void kernel_launch(void* const* d_in, const int* in_sizes, int n_in,
                              void* d_out, int out_size, void* d_ws, size_t ws_size,
                              hipStream_t stream) {
    (void)in_sizes; (void)n_in; (void)out_size; (void)ws_size;
    const float* hs  = (const float*)d_in[0];
    const int*   pos = (const int*)d_in[1];
    const float* Wq  = (const float*)d_in[2];
    const float* Wk  = (const float*)d_in[3];
    const float* Wv  = (const float*)d_in[4];
    const float* Wo  = (const float*)d_in[5];
    const float* qs  = (const float*)d_in[6];
    const float* ks  = (const float*)d_in[7];

    const size_t bhnd = (size_t)Bsz * Hh * Nsq * DH;   // 6.29M elems
    const size_t hsn  = (size_t)Bsz * Nsq * Dm;        // 6.29M elems
    __bf16* qb    = (__bf16*)d_ws;
    __bf16* kb    = qb + bhnd;
    __bf16* vt    = kb + bhnd;           // [B,H,DH,N]
    __bf16* attnb = vt + bhnd;           // [BN, 768]
    __bf16* hsb   = attnb + hsn;         // [BN, 768]
    __bf16* Wb    = hsb + hsn;           // Wq|Wk|Wv|Wo packed, 4*768*768

    dim3 blk(128);
    // stage 0: pack to bf16
    {
        int n4 = (int)(hsn / 4);
        pack_bf16_kernel<<<dim3((n4 + 255) / 256), dim3(256), 0, stream>>>(hs, hsb, n4);
        int w4 = WSZ / 4;
        dim3 g((w4 + 255) / 256), b256(256);
        pack_bf16_kernel<<<g, b256, 0, stream>>>(Wq, Wb + 0 * (size_t)WSZ, w4);
        pack_bf16_kernel<<<g, b256, 0, stream>>>(Wk, Wb + 1 * (size_t)WSZ, w4);
        pack_bf16_kernel<<<g, b256, 0, stream>>>(Wv, Wb + 2 * (size_t)WSZ, w4);
        pack_bf16_kernel<<<g, b256, 0, stream>>>(Wo, Wb + 3 * (size_t)WSZ, w4);
    }
    qkv_norm_rope_kernel<<<dim3((Bsz * Nsq) / 16, Hh, 3), blk, 0, stream>>>(
        hsb, Wb, pos, qs, ks, qb, kb, vt);
    attn_kernel<<<dim3(Nsq / 16, Hh, Bsz), blk, 0, stream>>>(qb, kb, vt, attnb);
    out_proj_kernel<<<dim3((Bsz * Nsq) / 16, Dm / DH, 1), blk, 0, stream>>>(
        attnb, Wb + 3 * (size_t)WSZ, (float*)d_out);
}
